// SO2_Linear_45586782880340
// MI455X (gfx1250) — matrix operand order repaired and checked by reading, not compiled, not run on hardware
//
#include <hip/hip_runtime.h>
#include <hip/hip_bf16.h>

typedef __attribute__((ext_vector_type(16))) _Float16 v16h;
typedef __attribute__((ext_vector_type(8)))  _Float16 v8h;
typedef __attribute__((ext_vector_type(8)))  float    v8f;

#define N_NODES   64000
#define DIMF      576
#define TILE      64          // nodes per block
#define THREADS   128         // 4 wave32
#define BLK_PER_B 125         // 8000 / 64
#define AST       584         // f16 stride of gathered-A rows (pad for banks, 16B aligned)
#define YST       584         // f32 stride of Y rows

// LDS layout (bytes)
#define JF_OFF    0           // 96 f32
#define D_OFF     384         // 2*64*81 f32 = 41472 B
#define A_OFF     41856       // 64*584 f16  = 74752 B
#define Y_OFF     116608      // 64*584 f32  = 149504 B
#define LDS_BYTES 266112

__constant__ int   c_ANTI[9] = {0, 3, 2, 1, 8, 7, 6, 5, 4};
__constant__ float c_FREQ[9] = {0.f, 1.f, 0.f, -1.f, 2.f, 1.f, 0.f, -1.f, -2.f};

// ---------- Wigner helpers (sparse z-rotation chains) ----------
__device__ inline void mul_row_Jf(const float* v, const float* Jf, float* t) {
#pragma unroll
  for (int c = 0; c < 9; ++c) {
    float s = 0.f;
#pragma unroll
    for (int r = 0; r < 9; ++r) s = fmaf(v[r], Jf[r * 9 + c], s);
    t[c] = s;
  }
}

// out = in @ X(ang);  X[r][r]=cos(f_r a), X[r][anti_r]=sin(f_r a)  (r != anti_r)
__device__ inline void z_rot_right(const float* in, float ang, float* out) {
#pragma unroll
  for (int c = 0; c < 9; ++c) {
    int ac = c_ANTI[c];
    float f = c_FREQ[c];
    if (ac == c) out[c] = in[c];
    else         out[c] = in[c] * cosf(f * ang) - in[ac] * sinf(f * ang);
  }
}

// row i of Xa @ Jf @ Xb @ Jf @ Xc
__device__ inline void wigner_row(const float* Jf, int i, float a, float b, float g,
                                  float* outrow) {
  float v[9], t[9];
  float fi = c_FREQ[i];
  int   ai = c_ANTI[i];
  float ca = cosf(fi * a), sa = sinf(fi * a);
#pragma unroll
  for (int k = 0; k < 9; ++k)
    v[k] = (k == i) ? ca : ((k == ai && ai != i) ? sa : 0.f);
  mul_row_Jf(v, Jf, t);
  z_rot_right(t, b, v);
  mul_row_Jf(v, Jf, t);
  z_rot_right(t, g, outrow);
}

// gathered m-order position inside a 576-wide row
__device__ inline int gpos(int l, int c, int i) {
  if (l == 0) return c;
  if (l == 1) return (i == 1) ? 64 + c : ((i == 0) ? 192 + c : 320 + c);
  switch (i) {                       // l == 2
    case 2:  return 128 + c;
    case 1:  return 256 + c;
    case 3:  return 384 + c;
    case 0:  return 448 + c;
    default: return 512 + c;
  }
}

// ---------- WMMA helpers ----------
__device__ inline v8f wmma16(v16h a, v16h b, v8f c) {
  return __builtin_amdgcn_wmma_f32_16x16x32_f16(false, a, false, b, (short)0, c,
                                                false, false);
}

// A fragment (16x32 f16): per-lane two contiguous 16B chunks at k0 and k0+8
__device__ inline v16h load_a_frag(const _Float16* p) {
  union { v16h v; v8h h[2]; } u;
  u.h[0] = *(const v8h*)(p);
  u.h[1] = *(const v8h*)(p + 8);
  return u.v;
}

// ======================= prep: fold experts per batch =======================
__global__ void moe_prep(const float* __restrict__ coef,
                         const float* __restrict__ w0,  const float* __restrict__ b0,
                         const float* __restrict__ ws0, const float* __restrict__ bs0,
                         const float* __restrict__ w1,  const float* __restrict__ ws1,
                         const float* __restrict__ w2,  const float* __restrict__ ws2,
                         _Float16* __restrict__ W0c, _Float16* __restrict__ W1c,
                         _Float16* __restrict__ W2c, float* __restrict__ Bias0) {
  const int b = blockIdx.x;
  float cf[8];
#pragma unroll
  for (int e = 0; e < 8; ++e) cf[e] = coef[b * 8 + e];

  for (int t = threadIdx.x; t < 192 * 192; t += blockDim.x) {
    float acc = ws0[t];
#pragma unroll
    for (int e = 0; e < 8; ++e) acc = fmaf(cf[e], w0[e * 36864 + t], acc);
    W0c[b * 36864 + t] = (_Float16)acc;
  }
  for (int t = threadIdx.x; t < 256 * 128; t += blockDim.x) {
    float acc = ws1[t];
#pragma unroll
    for (int e = 0; e < 8; ++e) acc = fmaf(cf[e], w1[e * 32768 + t], acc);
    W1c[b * 32768 + t] = (_Float16)acc;
  }
  for (int t = threadIdx.x; t < 128 * 64; t += blockDim.x) {
    float acc = ws2[t];
#pragma unroll
    for (int e = 0; e < 8; ++e) acc = fmaf(cf[e], w2[e * 8192 + t], acc);
    W2c[b * 8192 + t] = (_Float16)acc;
  }
  for (int t = threadIdx.x; t < 192; t += blockDim.x) {
    float acc = bs0[t];
#pragma unroll
    for (int e = 0; e < 8; ++e) acc = fmaf(cf[e], b0[e * 192 + t], acc);
    Bias0[b * 192 + t] = acc;
  }
}

// ======================= fused main kernel =======================
__global__ __launch_bounds__(THREADS, 1)
void so2_fused(const float* __restrict__ x,    const float* __restrict__ alpha,
               const float* __restrict__ beta, const float* __restrict__ gmma,
               const float* __restrict__ Jd1,  const float* __restrict__ Jd2,
               const _Float16* __restrict__ W0c, const _Float16* __restrict__ W1c,
               const _Float16* __restrict__ W2c, const float* __restrict__ Bias0,
               float* __restrict__ out) {
  extern __shared__ char smem[];
  float*    Jf   = (float*)(smem + JF_OFF);
  float*    Dm   = (float*)(smem + D_OFF);     // [2][TILE][81]
  _Float16* Alds = (_Float16*)(smem + A_OFF);  // [TILE][AST] gathered f16
  float*    Ylds = (float*)(smem + Y_OFF);     // [TILE][YST] original order f32

  const int tid  = threadIdx.x;
  const int lane = tid & 31;
  const int wv   = tid >> 5;        // 0..3
  const int lcol = lane & 15;
  const int lhi  = lane >> 4;       // 0 or 1
  const int bb   = blockIdx.x / BLK_PER_B;
  const int n0   = blockIdx.x * TILE;

  // ---- Phase 0a: assemble Jf ----
  if (tid < 81) {
    int r = tid / 9, c = tid % 9;
    float v = 0.f;
    if (r == 0 && c == 0)                       v = 1.f;
    else if (r >= 1 && r < 4 && c >= 1 && c < 4) v = Jd1[(r - 1) * 3 + (c - 1)];
    else if (r >= 4 && c >= 4)                   v = Jd2[(r - 4) * 5 + (c - 4)];
    Jf[r * 9 + c] = v;
  }
  __syncthreads();

  // ---- Phase 0b: forward + inverse Wigner matrices ----
  for (int t = tid; t < 2 * TILE * 9; t += THREADS) {
    int which = t / (TILE * 9);
    int rem   = t % (TILE * 9);
    int nl = rem / 9, i = rem % 9;
    int n = n0 + nl;
    float a = alpha[n], b = beta[n], g = gmma[n];
    float A0 = (which == 0) ? a : -g;
    float B0 = (which == 0) ? b : -b;
    float G0 = (which == 0) ? g : -a;
    wigner_row(Jf, i, A0, B0, G0, &Dm[(which * TILE + nl) * 81 + i * 9]);
  }
  __syncthreads();

  // ---- Phase 1: rotate x, write f16 into gathered layout ----
  for (int t = tid; t < TILE * 192; t += THREADS) {
    int nl = t / 192, j = t % 192;
    int l = j >> 6, c = j & 63;
    int d = 2 * l + 1, off = l * l;
    int col0 = (l == 0) ? 0 : ((l == 1) ? 64 : 256);
    const float* Df = &Dm[nl * 81];
    const float* xp = x + (size_t)(n0 + nl) * DIMF + col0 + c * d;
    float xv[5];
    for (int jj = 0; jj < d; ++jj) xv[jj] = xp[jj];
    for (int i = 0; i < d; ++i) {
      float s = 0.f;
      for (int jj = 0; jj < d; ++jj)
        s = fmaf(Df[(off + i) * 9 + off + jj], xv[jj], s);
      Alds[nl * AST + gpos(l, c, i)] = (_Float16)s;
    }
  }
  __syncthreads();

  // ---- Phase 2: three WMMA GEMMs ----
  // m0: M=64 (rows=nodes), K=192 @ offset 0, N=192, +bias
  {
    const int rt = wv;                       // row tile
    v16h afr[6];
#pragma unroll
    for (int ks = 0; ks < 6; ++ks) {
      int m  = rt * 16 + lcol;
      int k0 = ks * 32 + lhi * 16;
      afr[ks] = load_a_frag(&Alds[m * AST + k0]);
    }
    const _Float16* Wb = W0c + (size_t)bb * 36864;
    for (int ct = 0; ct < 12; ++ct) {
      v8f acc = {};
#pragma unroll
      for (int ks = 0; ks < 6; ++ks) {
        const _Float16* bp = Wb + (ct * 16 + lcol) * 192 + ks * 32 + lhi * 16;
        acc = wmma16(afr[ks], *(const v16h*)bp, acc);
      }
      int o = ct * 16 + lcol;
      int oc = (o < 64) ? o : ((o < 128) ? 64 + 3 * (o - 64) + 1
                                         : 256 + 5 * (o - 128) + 2);
      float bias = Bias0[bb * 192 + o];
#pragma unroll
      for (int r = 0; r < 8; ++r) {
        int nlr = rt * 16 + r + lhi * 8;
        Ylds[nlr * YST + oc] = acc[r] + bias;
      }
    }
  }
  // m1: M=128 (rows = node*2+p), K=128 @ 192+p*128, N=256, complex combine
  {
    const _Float16* Wb = W1c + (size_t)bb * 32768;
    for (int rr = 0; rr < 2; ++rr) {
      int rt = wv + rr * 4;
      v16h afr[4];
#pragma unroll
      for (int ks = 0; ks < 4; ++ks) {
        int m = rt * 16 + lcol;
        int node = m >> 1, p = m & 1;
        int k0 = 192 + p * 128 + ks * 32 + lhi * 16;
        afr[ks] = load_a_frag(&Alds[node * AST + k0]);
      }
      for (int cp = 0; cp < 8; ++cp) {
        v8f acc0 = {}, acc1 = {};
#pragma unroll
        for (int ks = 0; ks < 4; ++ks) {
          const _Float16* b0 = Wb + (cp * 16 + lcol) * 128 + ks * 32 + lhi * 16;
          const _Float16* b1 = Wb + ((cp + 8) * 16 + lcol) * 128 + ks * 32 + lhi * 16;
          acc0 = wmma16(afr[ks], *(const v16h*)b0, acc0);
          acc1 = wmma16(afr[ks], *(const v16h*)b1, acc1);
        }
        int k = cp * 16 + lcol;                      // 0..127
        int com = (k < 64) ? 64 + 3 * k : 256 + 5 * (k - 64) + 1;
        int cop = com + 2;
#pragma unroll
        for (int j = 0; j < 4; ++j) {
          int nl = rt * 8 + j + lhi * 4;
          float om = acc0[2 * j]     - acc1[2 * j + 1];
          float op = acc1[2 * j]     + acc0[2 * j + 1];
          Ylds[nl * YST + com] = om;
          Ylds[nl * YST + cop] = op;
        }
      }
    }
  }
  // m2: M=128, K=64 @ 448+p*64, N=128, complex combine
  {
    const _Float16* Wb = W2c + (size_t)bb * 8192;
    for (int rr = 0; rr < 2; ++rr) {
      int rt = wv + rr * 4;
      v16h afr[2];
#pragma unroll
      for (int ks = 0; ks < 2; ++ks) {
        int m = rt * 16 + lcol;
        int node = m >> 1, p = m & 1;
        int k0 = 448 + p * 64 + ks * 32 + lhi * 16;
        afr[ks] = load_a_frag(&Alds[node * AST + k0]);
      }
      for (int cp = 0; cp < 4; ++cp) {
        v8f acc0 = {}, acc1 = {};
#pragma unroll
        for (int ks = 0; ks < 2; ++ks) {
          const _Float16* b0 = Wb + (cp * 16 + lcol) * 64 + ks * 32 + lhi * 16;
          const _Float16* b1 = Wb + ((cp + 4) * 16 + lcol) * 64 + ks * 32 + lhi * 16;
          acc0 = wmma16(afr[ks], *(const v16h*)b0, acc0);
          acc1 = wmma16(afr[ks], *(const v16h*)b1, acc1);
        }
        int k = cp * 16 + lcol;                      // 0..63
        int com = 256 + 5 * k;
        int cop = com + 4;
#pragma unroll
        for (int j = 0; j < 4; ++j) {
          int nl = rt * 8 + j + lhi * 4;
          float om = acc0[2 * j]     - acc1[2 * j + 1];
          float op = acc1[2 * j]     + acc0[2 * j + 1];
          Ylds[nl * YST + com] = om;
          Ylds[nl * YST + cop] = op;
        }
      }
    }
  }
  __syncthreads();

  // ---- Phase 3: inverse rotation, write out ----
  for (int t = tid; t < TILE * 192; t += THREADS) {
    int nl = t / 192, j = t % 192;
    int l = j >> 6, c = j & 63;
    int d = 2 * l + 1, off = l * l;
    int col0 = (l == 0) ? 0 : ((l == 1) ? 64 : 256);
    int base = col0 + c * d;
    const float* Di = &Dm[(TILE + nl) * 81];
    float xv[5];
    for (int jj = 0; jj < d; ++jj) xv[jj] = Ylds[nl * YST + base + jj];
    for (int i = 0; i < d; ++i) {
      float s = 0.f;
      for (int jj = 0; jj < d; ++jj)
        s = fmaf(Di[(off + i) * 9 + off + jj], xv[jj], s);
      out[(size_t)(n0 + nl) * DIMF + base + i] = s;
    }
  }
}

// ======================= host launch =======================
extern "C" void kernel_launch(void* const* d_in, const int* in_sizes, int n_in,
                              void* d_out, int out_size, void* d_ws, size_t ws_size,
                              hipStream_t stream) {
  (void)in_sizes; (void)n_in; (void)out_size; (void)ws_size;
  const float* x     = (const float*)d_in[0];
  const float* alpha = (const float*)d_in[1];
  const float* beta  = (const float*)d_in[2];
  const float* gmma  = (const float*)d_in[3];
  const float* coef  = (const float*)d_in[4];
  const float* Jd1   = (const float*)d_in[5];
  const float* Jd2   = (const float*)d_in[6];
  const float* w0    = (const float*)d_in[7];
  const float* b0    = (const float*)d_in[8];
  const float* ws0   = (const float*)d_in[9];
  const float* bs0   = (const float*)d_in[10];
  const float* w1    = (const float*)d_in[11];
  const float* ws1   = (const float*)d_in[12];
  const float* w2    = (const float*)d_in[13];
  const float* ws2   = (const float*)d_in[14];
  float* out = (float*)d_out;

  char* ws = (char*)d_ws;
  _Float16* W0c   = (_Float16*)(ws);
  _Float16* W1c   = (_Float16*)(ws + 8 * 36864 * 2);
  _Float16* W2c   = (_Float16*)(ws + 8 * 36864 * 2 + 8 * 32768 * 2);
  float*    Bias0 = (float*)   (ws + 8 * 36864 * 2 + 8 * 32768 * 2 + 8 * 8192 * 2);

  hipFuncSetAttribute((const void*)so2_fused,
                      hipFuncAttributeMaxDynamicSharedMemorySize, LDS_BYTES);

  moe_prep<<<8, 256, 0, stream>>>(coef, w0, b0, ws0, bs0, w1, ws1, w2, ws2,
                                  W0c, W1c, W2c, Bias0);
  so2_fused<<<N_NODES / TILE, THREADS, LDS_BYTES, stream>>>(
      x, alpha, beta, gmma, Jd1, Jd2, W0c, W1c, W2c, Bias0, out);
}
